// MultiHeadAttention_29515015258680
// MI455X (gfx1250) — compile-verified, matrix operand order checked by reading, not run on hardware
//
#include <hip/hip_runtime.h>
#include <cstdint>
#include <cstddef>

// MultiHeadAttention: B=4, S=2048, D=1024, H=16, DK=64
#define BB 4
#define SS 2048
#define DD 1024
#define HH 16
#define DKK 64

typedef __attribute__((ext_vector_type(16))) __bf16 v16bf;
typedef __attribute__((ext_vector_type(8)))  float  v8f;

// TDM builtin arg types (amdgpu-toolchain clang-23 6-arg form)
typedef __attribute__((ext_vector_type(4))) unsigned int tdm_u32x4;
typedef __attribute__((ext_vector_type(8))) int          tdm_i32x8;
typedef __attribute__((ext_vector_type(4))) int          tdm_i32x4;

// float -> bf16 round-to-nearest-even (bit ops)
__device__ __forceinline__ __bf16 f2bf(float f) {
  union { float f; uint32_t u; } a; a.f = f;
  uint32_t r = (a.u + 0x7FFFu + ((a.u >> 16) & 1u)) >> 16;
  union { uint16_t s; __bf16 b; } o; o.s = (uint16_t)r;
  return o.b;
}

// pack two floats into one u32 of two bf16 (RNE)
__device__ __forceinline__ uint32_t pk2bf(float lo, float hi) {
  union { float f; uint32_t u; } a, b; a.f = lo; b.f = hi;
  uint32_t rl = (a.u + 0x7FFFu + ((a.u >> 16) & 1u)) >> 16;
  uint32_t rh = (b.u + 0x7FFFu + ((b.u >> 16) & 1u)) & 0xFFFF0000u;
  return rh | (rl & 0xFFFFu);
}

// Load one 16x32 bf16 fragment from a K-contiguous 32-element row
// (ISA 7.12.2: two contiguous 8-element runs -> two b128 loads).
__device__ __forceinline__ v16bf ld_frag(const __bf16* row, int half) {
  union { uint4 q[2]; v16bf v; } u;
  const uint4* q = (const uint4*)row;
  u.q[0] = q[half];
  u.q[1] = q[2 + half];
  return u.v;
}

__device__ __forceinline__ v8f wmma_bf16(v16bf a, v16bf b, v8f c) {
  return __builtin_amdgcn_wmma_f32_16x16x32_bf16(
      false, a, false, b, (short)0, c, false, false);
}

// ---------------------------------------------------------------------------
// Pre-pass A: elementwise fp32 -> bf16 (8 elems/thread, b128 in/out)
// ---------------------------------------------------------------------------
__global__ __launch_bounds__(256) void cvt_bf16_kernel(
    const float* __restrict__ in, __bf16* __restrict__ outp)
{
  const size_t i = (size_t)blockIdx.x * 256 + threadIdx.x;
  const float4* p = (const float4*)(in + i * 8);
  const float4 a = p[0], b = p[1];
  uint4 o;
  o.x = pk2bf(a.x, a.y); o.y = pk2bf(a.z, a.w);
  o.z = pk2bf(b.x, b.y); o.w = pk2bf(b.z, b.w);
  *(uint4*)(outp + i * 8) = o;
}

// ---------------------------------------------------------------------------
// Pre-pass B: Wq/Wk/Wv [h][k][n] fp32 -> Wt [z][h*64+n][k] bf16 (transposed,
// so GEMM B-fragments are K-contiguous b128 loads straight from memory).
// ---------------------------------------------------------------------------
__global__ __launch_bounds__(256) void cvt_w_transpose_kernel(
    const float* __restrict__ Wq, const float* __restrict__ Wk,
    const float* __restrict__ Wv, __bf16* __restrict__ Wt)
{
  const int z = blockIdx.y;
  const float* Wp = (z == 0) ? Wq : (z == 1) ? Wk : Wv;
  const int row = blockIdx.x;            // 0..1023 = h*64 + n
  const int h = row >> 6, n = row & 63;
  const int k = threadIdx.x * 4;         // 256 threads x 4 k = 1024
  const float* src = Wp + (size_t)h * DD * DKK + (size_t)k * DKK + n;
  const float f0 = src[0], f1 = src[DKK], f2 = src[2 * DKK], f3 = src[3 * DKK];
  uint2 o; o.x = pk2bf(f0, f1); o.y = pk2bf(f2, f3);
  *(uint2*)(Wt + (size_t)z * (DD * DD) + (size_t)row * DD + k) = o;
}

// ---------------------------------------------------------------------------
// Kernel 1: QKV projection GEMM. M=32 tile, N=DK per block (4 waves x 16n).
// All operands pre-formatted bf16, K-contiguous -> no LDS, pure load+WMMA.
// Weights/x are L2-resident (8MB + 16MB vs 192MB L2).
// ---------------------------------------------------------------------------
__global__ __launch_bounds__(128) void qkv_gemm_kernel(
    const __bf16* __restrict__ xbf, const __bf16* __restrict__ Wt,
    const float* __restrict__ bq, const float* __restrict__ bk,
    const float* __restrict__ bv, __bf16* __restrict__ Qall)
{
  const int z = blockIdx.z;
  const float* bp = (z == 0) ? bq : (z == 1) ? bk : bv;
  __bf16* Op = Qall + (size_t)z * ((size_t)BB * HH * SS * DKK);

  const int h    = blockIdx.y;
  const int r0   = blockIdx.x * 32;          // row over flattened B*S
  const int wave = threadIdx.x >> 5;
  const int lane = threadIdx.x & 31;
  const int half = lane >> 4;
  const int lm   = lane & 15;
  const int n0   = wave * 16;

  const __bf16* arow0 = xbf + (size_t)(r0 + lm) * DD;
  const __bf16* arow1 = xbf + (size_t)(r0 + 16 + lm) * DD;
  const __bf16* brow  = Wt + (size_t)z * (DD * DD)
                           + (size_t)(h * DKK + n0 + lm) * DD;

  v8f acc0 = {}, acc1 = {};
  #pragma unroll 4
  for (int k0 = 0; k0 < DD; k0 += 32) {
    const v16bf bfm = ld_frag(brow + k0, half);
    const v16bf af0 = ld_frag(arow0 + k0, half);
    const v16bf af1 = ld_frag(arow1 + k0, half);
    acc0 = wmma_bf16(af0, bfm, acc0);
    acc1 = wmma_bf16(af1, bfm, acc1);
  }

  const int b  = r0 >> 11;
  const int s0 = r0 & (SS - 1);
  const float bias = bp[h * DKK + n0 + lm];
  #pragma unroll
  for (int r = 0; r < 8; r++) {
    const int m0 = r + half * 8;             // C layout: M = vgpr + 8*half
    Op[((size_t)(b * HH + h) * SS + s0 + m0) * DKK + n0 + lm] = f2bf(acc0[r] + bias);
    Op[((size_t)(b * HH + h) * SS + s0 + 16 + m0) * DKK + n0 + lm] = f2bf(acc1[r] + bias);
  }
}

// ---------------------------------------------------------------------------
// Kernel 2: causal flash attention, one wave per 16-row Q tile.
// K tile staged by the Tensor Data Mover; V staged transposed [dk][t32].
// ---------------------------------------------------------------------------
__global__ __launch_bounds__(32) void flash_attn_kernel(
    const __bf16* __restrict__ Qg, const __bf16* __restrict__ Kg,
    const __bf16* __restrict__ Vg, __bf16* __restrict__ Og)
{
  const int mt   = blockIdx.x;
  const int h    = blockIdx.y;
  const int b    = blockIdx.z;
  const int bh   = b * HH + h;
  const int lane = threadIdx.x;
  const int half = lane >> 4;
  const int lm   = lane & 15;
  const int s0   = mt * 16;
  const float scale = 0.125f;               // 1/sqrt(64)

  __shared__ __align__(16) __bf16 sK[32 * 64];    // [t][dk] row-major
  __shared__ __align__(16) __bf16 sVt[64 * 32];   // [dk][t]  transposed
  __shared__ __align__(16) __bf16 sP[16 * 32];    // [m][t32]

  v16bf qa[2];
  const __bf16* qrow = Qg + ((size_t)bh * SS + s0 + lm) * DKK;
  qa[0] = ld_frag(qrow, half);
  qa[1] = ld_frag(qrow + 32, half);

  float rm[8], rl[8];
  v8f acc[4] = {};
  #pragma unroll
  for (int r = 0; r < 8; r++) { rm[r] = -__builtin_inff(); rl[r] = 0.f; }

  const int tp = lane & 15;   // V-stage: t rows 2tp,2tp+1
  const int dh = lane >> 4;   // V-stage: dk block [32dh, 32dh+32)

  const int tmax = s0 + 16;
  for (int t0 = 0; t0 < tmax; t0 += 32) {
    // ---- K tile [32 x 64] via Tensor Data Mover ----
#if __has_builtin(__builtin_amdgcn_tensor_load_to_lds) && __has_builtin(__builtin_amdgcn_s_wait_tensorcnt)
    {
      const uint64_t ga = (uint64_t)(uintptr_t)(Kg + ((size_t)bh * SS + t0) * DKK);
      const uint32_t la = (uint32_t)(uintptr_t)&sK[0];
      tdm_u32x4 g0;
      g0[0] = 1u;                                   // count=1, user mode
      g0[1] = la;                                   // lds_addr
      g0[2] = (uint32_t)ga;                         // global_addr[31:0]
      g0[3] = ((uint32_t)(ga >> 32) & 0x01FFFFFFu)  // global_addr[56:32]
              | 0x80000000u;                        // type=2 ("image")
      tdm_i32x8 g1;
      g1[0] = 0x00010000;                     // wg_mask=0, data_size=2 bytes
      g1[1] = (int)(64u << 16);               // tensor_dim0=64
      g1[2] = (int)((2048u & 0xFFFFu) << 16); // tensor_dim1 lo16
      g1[3] = (int)(64u << 16);               // tile_dim0=64
      g1[4] = 32;                             // tile_dim1=32
      g1[5] = 64;                             // tensor_dim0_stride=64
      g1[6] = 0; g1[7] = 0;
      tdm_i32x4 gz;  gz[0] = 0; gz[1] = 0; gz[2] = 0; gz[3] = 0;
      tdm_i32x8 gz8; gz8[0] = 0; gz8[1] = 0; gz8[2] = 0; gz8[3] = 0;
      gz8[4] = 0; gz8[5] = 0; gz8[6] = 0; gz8[7] = 0;
      __builtin_amdgcn_tensor_load_to_lds(g0, g1, gz, gz, gz8, 0);
      __builtin_amdgcn_s_wait_tensorcnt(0);
    }
#else
    {
      const uint4* kr = (const uint4*)(Kg + ((size_t)bh * SS + t0 + lane) * DKK);
      uint4* dk4 = (uint4*)sK;
      #pragma unroll
      for (int i = 0; i < 8; i++) dk4[lane * 8 + i] = kr[i];
    }
#endif

    // ---- V tile staged transposed: sVt[dk][t] with packed 2x2 transposes
    {
      const uint4* v0p = (const uint4*)(Vg + ((size_t)bh * SS + t0 + 2 * tp) * DKK + dh * 32);
      const uint4* v1p = v0p + 8;   // next t row (64 bf16 = 8 uint4)
      #pragma unroll
      for (int c = 0; c < 4; c++) {
        const uint4 r0v = v0p[c], r1v = v1p[c];
        const uint32_t w0[4] = {r0v.x, r0v.y, r0v.z, r0v.w};
        const uint32_t w1[4] = {r1v.x, r1v.y, r1v.z, r1v.w};
        #pragma unroll
        for (int w = 0; w < 4; w++) {
          const int dk = dh * 32 + c * 8 + w * 2;
          const uint32_t even = (w1[w] << 16) | (w0[w] & 0xFFFFu);
          const uint32_t odd  = (w1[w] & 0xFFFF0000u) | (w0[w] >> 16);
          *(uint32_t*)&sVt[(size_t)dk * 32 + 2 * tp]       = even;
          *(uint32_t*)&sVt[(size_t)(dk + 1) * 32 + 2 * tp] = odd;
        }
      }
    }
    __syncthreads();

    // ---- scores: two 16x16 tiles (dk-contiguous fragment rows in sK)
    v8f sc[2] = {};
    #pragma unroll
    for (int ti = 0; ti < 2; ti++) {
      #pragma unroll
      for (int kc = 0; kc < 2; kc++) {
        const v16bf kb = ld_frag(&sK[(ti * 16 + lm) * 64 + kc * 32], half);
        sc[ti] = wmma_bf16(qa[kc], kb, sc[ti]);
      }
    }

    // ---- online softmax over the 32 columns
    float e0[8], e1[8];
    #pragma unroll
    for (int r = 0; r < 8; r++) {
      const int qr = s0 + r + half * 8;
      float v0 = (t0 + lm      <= qr) ? sc[0][r] * scale : -__builtin_inff();
      float v1 = (t0 + 16 + lm <= qr) ? sc[1][r] * scale : -__builtin_inff();
      float mx = fmaxf(v0, v1);
      #pragma unroll
      for (int off = 1; off < 16; off <<= 1)
        mx = fmaxf(mx, __shfl_xor(mx, off, 32));
      const float nm   = fmaxf(rm[r], mx);
      const float corr = __expf(rm[r] - nm);
      const float a0   = __expf(v0 - nm);
      const float a1   = __expf(v1 - nm);
      float sum = a0 + a1;
      #pragma unroll
      for (int off = 1; off < 16; off <<= 1)
        sum += __shfl_xor(sum, off, 32);
      rl[r] = rl[r] * corr + sum;
      rm[r] = nm;
      #pragma unroll
      for (int nt = 0; nt < 4; nt++) acc[nt][r] = acc[nt][r] * corr;
      e0[r] = a0; e1[r] = a1;
    }

    // ---- P (C-layout) -> LDS [m][t32]
    #pragma unroll
    for (int r = 0; r < 8; r++) {
      sP[(r + half * 8) * 32 + lm]      = f2bf(e0[r]);
      sP[(r + half * 8) * 32 + 16 + lm] = f2bf(e1[r]);
    }
    __syncthreads();

    const v16bf pa = ld_frag(&sP[lm * 32], half);
    #pragma unroll
    for (int nt = 0; nt < 4; nt++) {
      const v16bf vb = ld_frag(&sVt[(nt * 16 + lm) * 32], half);
      acc[nt] = wmma_bf16(pa, vb, acc[nt]);
    }
    __syncthreads();
  }

  #pragma unroll
  for (int nt = 0; nt < 4; nt++)
    #pragma unroll
    for (int r = 0; r < 8; r++) {
      const float o = acc[nt][r] / rl[r];
      Og[((size_t)bh * SS + s0 + r + half * 8) * DKK + nt * 16 + lm] = f2bf(o);
    }
}

// ---------------------------------------------------------------------------
// Kernel 3: output projection GEMM. out[m,n] = sum_k concat[m,k]*Wo[n,k]+bo.
// A from bf16 O (head-aware addressing, 32-chunks never cross a head);
// B from pre-converted bf16 Wot [n][k]. No LDS, pure load+WMMA.
// ---------------------------------------------------------------------------
__global__ __launch_bounds__(128) void out_gemm_kernel(
    const __bf16* __restrict__ Og, const __bf16* __restrict__ Wot,
    const float* __restrict__ bo, float* __restrict__ out)
{
  const int wave = threadIdx.x >> 5;
  const int lane = threadIdx.x & 31;
  const int half = lane >> 4;
  const int lm   = lane & 15;
  const int r0   = blockIdx.x * 32;
  const int n0   = blockIdx.y * 64 + wave * 16;

  const int b  = r0 >> 11;
  const int s0 = r0 & (SS - 1);
  // per-lane A row bases inside O[b,h,s,dk]; head selected per k-chunk
  const __bf16* abase0 = Og + ((size_t)b * HH * SS + (s0 + lm)) * DKK;
  const __bf16* abase1 = Og + ((size_t)b * HH * SS + (s0 + 16 + lm)) * DKK;
  const __bf16* brow   = Wot + (size_t)(n0 + lm) * DD;

  v8f acc0 = {}, acc1 = {};
  #pragma unroll 4
  for (int k0 = 0; k0 < DD; k0 += 32) {
    const size_t hoff = (size_t)(k0 >> 6) * (SS * DKK) + (k0 & 63);
    const v16bf bfm = ld_frag(brow + k0, half);
    const v16bf af0 = ld_frag(abase0 + hoff, half);
    const v16bf af1 = ld_frag(abase1 + hoff, half);
    acc0 = wmma_bf16(af0, bfm, acc0);
    acc1 = wmma_bf16(af1, bfm, acc1);
  }

  const float bias = bo[n0 + lm];
  #pragma unroll
  for (int r = 0; r < 8; r++) {
    const int m0 = r0 + r + half * 8;
    out[(size_t)m0 * DD + n0 + lm]        = acc0[r] + bias;
    out[(size_t)(m0 + 16) * DD + n0 + lm] = acc1[r] + bias;
  }
}

// ---------------------------------------------------------------------------
extern "C" void kernel_launch(void* const* d_in, const int* in_sizes, int n_in,
                              void* d_out, int out_size, void* d_ws, size_t ws_size,
                              hipStream_t stream) {
  const float* x  = (const float*)d_in[0];
  const float* Wq = (const float*)d_in[1];
  const float* bq = (const float*)d_in[2];
  const float* Wk = (const float*)d_in[3];
  const float* bk = (const float*)d_in[4];
  const float* Wv = (const float*)d_in[5];
  const float* bv = (const float*)d_in[6];
  const float* Wo = (const float*)d_in[7];
  const float* bo = (const float*)d_in[8];
  float* out = (float*)d_out;

  // workspace layout (bf16):
  //   Q,K,V,O   : 4 x 8,388,608            (64 MB)
  //   xbf       : 8,388,608                (16 MB)
  //   Wt (qkv)  : 3 x 1,048,576            ( 6 MB)
  //   Wot       : 1,048,576                ( 2 MB)
  const size_t tsz = (size_t)BB * HH * SS * DKK;   // 8,388,608
  __bf16* Qw  = (__bf16*)d_ws;
  __bf16* Kw  = Qw + tsz;
  __bf16* Vw  = Kw + tsz;
  __bf16* Ow  = Vw + tsz;
  __bf16* xbf = Ow + tsz;
  __bf16* Wt  = xbf + tsz;
  __bf16* Wot = Wt + (size_t)3 * DD * DD;

  // pre-passes: one-time bf16 formatting
  cvt_bf16_kernel<<<dim3((unsigned)(tsz / 8 / 256)), 256, 0, stream>>>(x, xbf);
  cvt_w_transpose_kernel<<<dim3(HH * DKK, 3), 256, 0, stream>>>(Wq, Wk, Wv, Wt);
  cvt_bf16_kernel<<<dim3((unsigned)((size_t)DD * DD / 8 / 256)), 256, 0, stream>>>(Wo, Wot);

  // compute
  qkv_gemm_kernel<<<dim3((BB * SS) / 32, HH, 3), 128, 0, stream>>>(
      xbf, Wt, bq, bk, bv, Qw);
  flash_attn_kernel<<<dim3(SS / 16, HH, BB), 32, 0, stream>>>(Qw, Kw, Vw, Ow);
  out_gemm_kernel<<<dim3((BB * SS) / 32, DD / 64), 128, 0, stream>>>(
      Ow, Wot, bo, out);
}
